// CausalSelfAttention_39462159516054
// MI455X (gfx1250) — matrix operand order repaired and checked by reading, not compiled, not run on hardware
//
#include <hip/hip_runtime.h>
#include <hip/hip_bf16.h>
#include <math.h>

// ---------------------------------------------------------------------------
// CDNA5 / gfx1250 wave32 WMMA attention pipeline
// ---------------------------------------------------------------------------

typedef __bf16 bf16;
typedef __attribute__((ext_vector_type(16))) __bf16 v16bf;
typedef __attribute__((ext_vector_type(8)))  float  v8f;

#define SEQ     4096
#define DIMX    2048
#define NHEADS  16
#define HD      128          // head dim
#define HTOT    2048         // NHEADS*HD
#define QKVN    6144         // 3*HTOT
#define BR      64           // query rows per block (attention)
#define BC      64           // key cols per iteration (attention)
#define KP      40           // LDS pitch for 32-deep K tiles (80B, 16B-aligned frags)

__device__ __forceinline__ bf16 to_bf16(float x) { return (bf16)x; }
__device__ __forceinline__ bf16 to_bf16(bf16 x)  { return x; }

__device__ __forceinline__ unsigned pack_bf16x2(float a, float b) {
  union { bf16 h[2]; unsigned u; } cv;
  cv.h[0] = (bf16)a; cv.h[1] = (bf16)b;
  return cv.u;
}

// A-fragment (16x32 bf16, MxK): lane L holds row (L&15).
// half = L>>4. elems 0..15 -> K = k0 + half*16 + j (contiguous).
template <typename AT>
__device__ __forceinline__ v16bf load_a_frag(const AT* __restrict__ A, int lda,
                                             int row, int k0, int half) {
  const AT* p0 = A + (size_t)row * lda + k0 + half * 16;
  const AT* p1 = p0 + 8;
  v16bf a;
#pragma unroll
  for (int j = 0; j < 8; ++j) { a[j] = to_bf16(p0[j]); a[j + 8] = to_bf16(p1[j]); }
  return a;
}

// B-fragment (32x16 bf16, KxN): lane L holds column (L&15), K = half*16 + j.
// Source: 16 contiguous bf16 starting at the lane's (column-row, half*16).
__device__ __forceinline__ v16bf load_b_frag_bf16(const bf16* __restrict__ p) {
  v16bf b;
#pragma unroll
  for (int j = 0; j < 16; ++j) b[j] = p[j];
  return b;
}

__device__ __forceinline__ v8f wmma_bf16(v16bf a, v16bf b, v8f c) {
  return __builtin_amdgcn_wmma_f32_16x16x32_bf16(false, a, false, b, (short)0, c,
                                                 false, false);
}

// stage 16 contiguous source elems (fp32 or bf16) into LDS as bf16 (2x uint4)
__device__ __forceinline__ void stage16(const float* __restrict__ src,
                                        bf16* __restrict__ dst) {
  const float4 f0 = ((const float4*)src)[0];
  const float4 f1 = ((const float4*)src)[1];
  const float4 f2 = ((const float4*)src)[2];
  const float4 f3 = ((const float4*)src)[3];
  uint4 w0, w1;
  w0.x = pack_bf16x2(f0.x, f0.y); w0.y = pack_bf16x2(f0.z, f0.w);
  w0.z = pack_bf16x2(f1.x, f1.y); w0.w = pack_bf16x2(f1.z, f1.w);
  w1.x = pack_bf16x2(f2.x, f2.y); w1.y = pack_bf16x2(f2.z, f2.w);
  w1.z = pack_bf16x2(f3.x, f3.y); w1.w = pack_bf16x2(f3.z, f3.w);
  ((uint4*)dst)[0] = w0;
  ((uint4*)dst)[1] = w1;
}
__device__ __forceinline__ void stage16(const bf16* __restrict__ src,
                                        bf16* __restrict__ dst) {
  ((uint4*)dst)[0] = ((const uint4*)src)[0];
  ((uint4*)dst)[1] = ((const uint4*)src)[1];
}

// ---------------------------------------------------------------------------
// C[M,N] = A[M,K] * W[N,K]^T  (bf16 WMMA, fp32 accumulate), LDS-tiled.
// Block = 256 threads = 8 waves -> 128x128 C tile; wave = 32x64 tile.
// Per K-step of 32: stage As[128][32] + Bs[128][32] once, read frags from LDS.
// ---------------------------------------------------------------------------
template <typename AT, bool OUT_BF16>
__global__ __launch_bounds__(256) void gemm_wmma_kernel(
    const AT* __restrict__ A, const float* __restrict__ W, void* __restrict__ Cout,
    int M, int N, int K) {
  __shared__ bf16 As[128][KP];
  __shared__ bf16 Bs[128][KP];

  const int tid = threadIdx.x;
  const int lane = tid & 31;
  const int half = lane >> 4;
  const int lrow = lane & 15;
  const int wv = tid >> 5;       // 0..7
  const int mwv = wv & 3;        // 4 m-subtiles of 32 rows
  const int nwv = wv >> 2;       // 2 n-subtiles of 64 cols

  const int nblocks = N >> 7;
  const int mb = blockIdx.x / nblocks;
  const int nb = blockIdx.x - mb * nblocks;
  const int mbase = mb << 7;
  const int nbase = nb << 7;

  // staging role: row r = tid>>1 (0..127), 16-elem chunk c0 = (tid&1)*16
  const int sr = tid >> 1;
  const int sc = (tid & 1) * 16;

  v8f acc[2][4] = {};

  for (int k0 = 0; k0 < K; k0 += 32) {
    // ---- stage A and B tiles (fp32 -> bf16 conversion happens once) ------
    const AT* aSrc = A + (size_t)(mbase + sr) * K + k0 + sc;
    const float* bSrc = W + (size_t)(nbase + sr) * K + k0 + sc;
    if (k0 + 32 < K) {
      __builtin_prefetch(aSrc + 32, 0, 3);   // global_prefetch_b8 next K chunk
      __builtin_prefetch(bSrc + 32, 0, 3);
    }
    stage16(aSrc, &As[sr][sc]);
    stage16(bSrc, &Bs[sr][sc]);
    __syncthreads();

    // ---- 2 m-subtiles x 4 n-subtiles = 8 WMMAs per wave ------------------
#pragma unroll
    for (int sm = 0; sm < 2; ++sm) {
      v16bf a = load_a_frag(&As[0][0], KP, mwv * 32 + sm * 16 + lrow, 0, half);
#pragma unroll
      for (int n = 0; n < 4; ++n) {
        v16bf b = load_b_frag_bf16(&Bs[nwv * 64 + n * 16 + lrow][half * 16]);
        acc[sm][n] = wmma_bf16(a, b, acc[sm][n]);
      }
    }
    __syncthreads();
  }

  // C layout: vgpr r, lane L -> row = r + 8*half, col = lane&15 (per 16-col tile)
#pragma unroll
  for (int sm = 0; sm < 2; ++sm)
#pragma unroll
    for (int n = 0; n < 4; ++n)
#pragma unroll
      for (int r = 0; r < 8; ++r) {
        const size_t row = mbase + mwv * 32 + sm * 16 + half * 8 + r;
        const size_t col = nbase + nwv * 64 + n * 16 + lrow;
        if constexpr (OUT_BF16)
          ((bf16*)Cout)[row * N + col] = (bf16)acc[sm][n][r];
        else
          ((float*)Cout)[row * N + col] = acc[sm][n][r];
      }
}

// ---------------------------------------------------------------------------
// RMS-norm(q,k) + RoPE + v = l0*v + l1*ve.  One 128-thread block per (t,h).
// Writes Q/K/V as bf16 in (H, T, D) layout for the attention kernel.
// ---------------------------------------------------------------------------
__global__ __launch_bounds__(128) void norm_rope_mix_kernel(
    const bf16* __restrict__ qkv, const float* __restrict__ ve,
    const float* __restrict__ lambdas, bf16* __restrict__ Q, bf16* __restrict__ Ko,
    bf16* __restrict__ Vo) {
  __shared__ float red[128];
  __shared__ float qn[128];
  __shared__ float kn[128];

  const int t = blockIdx.x >> 4;
  const int h = blockIdx.x & (NHEADS - 1);
  const int d = threadIdx.x;

  const size_t rowb = (size_t)t * QKVN + h * HD;
  const float qv = (float)qkv[rowb + d];
  const float kv = (float)qkv[rowb + HTOT + d];
  const float vv = (float)qkv[rowb + 2 * HTOT + d];
  const float vev = ve[(size_t)t * DIMX + h * HD + d];
  const float eps = 1.1920929e-07f;

  red[d] = qv * qv;
  __syncthreads();
  for (int s = 64; s > 0; s >>= 1) {
    if (d < s) red[d] += red[d + s];
    __syncthreads();
  }
  const float rq = rsqrtf(red[0] * (1.0f / 128.0f) + eps);
  __syncthreads();

  red[d] = kv * kv;
  __syncthreads();
  for (int s = 64; s > 0; s >>= 1) {
    if (d < s) red[d] += red[d + s];
    __syncthreads();
  }
  const float rk = rsqrtf(red[0] * (1.0f / 128.0f) + eps);
  __syncthreads();

  qn[d] = qv * rq;
  kn[d] = kv * rk;
  __syncthreads();

  const size_t outb = (size_t)h * SEQ * HD + (size_t)t * HD;
  if (d < 64) {
    // dim_quarter = 32; freq j<32: (1/1024)^(j/31), j>=32: 0
    const float f = (d < 32) ? __powf(1.0f / 1024.0f, (float)d * (1.0f / 31.0f)) : 0.0f;
    const float th = (float)t * f;
    const float c = __cosf(th), s = __sinf(th);
    const float q1 = qn[d], q2 = qn[d + 64];
    Q[outb + d]      = (bf16)(q1 * c + q2 * s);
    Q[outb + d + 64] = (bf16)(-q1 * s + q2 * c);
    const float k1 = kn[d], k2 = kn[d + 64];
    Ko[outb + d]      = (bf16)(k1 * c + k2 * s);
    Ko[outb + d + 64] = (bf16)(-k1 * s + k2 * c);
  }
  Vo[outb + d] = (bf16)(lambdas[0] * vv + lambdas[1] * vev);
}

// ---------------------------------------------------------------------------
// Flash attention (causal). One block = one head x 64 query rows.
// 4 waves x 16 query rows each. K tile + V^T tile staged in LDS.
// ---------------------------------------------------------------------------
__global__ __launch_bounds__(128) void attention_kernel(
    const bf16* __restrict__ Q, const bf16* __restrict__ Kb,
    const bf16* __restrict__ Vb, bf16* __restrict__ Y) {
  __shared__ bf16 Kt[BC][HD + 8];        // (key, d)   row-major
  __shared__ bf16 Vt[HD][BC + 8];        // (d, key)   transposed
  __shared__ bf16 Ps[4][16][BC + 8];     // per-wave P staging (q-row, key)

  const int tid = threadIdx.x;
  const int wv = tid >> 5;
  const int lane = tid & 31;
  const int half = lane >> 4;
  const int lrow = lane & 15;
  const int qb = blockIdx.x >> 4;
  const int h = blockIdx.x & (NHEADS - 1);

  const bf16* Qh = Q + (size_t)h * SEQ * HD;
  const bf16* Kh = Kb + (size_t)h * SEQ * HD;
  const bf16* Vh = Vb + (size_t)h * SEQ * HD;

  // load this wave's Q fragments once (16 rows x 128 cols -> 4 A frags)
  v16bf aq[4];
#pragma unroll
  for (int kc = 0; kc < 4; ++kc)
    aq[kc] = load_a_frag(Qh, HD, qb * BR + wv * 16 + lrow, kc * 32, half);

  v8f o[8] = {};
  float m[8], l[8];
#pragma unroll
  for (int r = 0; r < 8; ++r) { m[r] = -INFINITY; l[r] = 0.0f; }

  const float scale = 0.08838834764831845f;  // 1/sqrt(128)

  for (int kb = 0; kb <= qb; ++kb) {
    // -- cooperative load of K tile and transposed V tile ------------------
    {
      const int r = tid >> 1;
      const int c0 = (tid & 1) * 64;
      const uint4* ksrc = (const uint4*)(Kh + (size_t)(kb * BC + r) * HD + c0);
      uint4* kdst = (uint4*)(&Kt[r][c0]);
#pragma unroll
      for (int j = 0; j < 8; ++j) kdst[j] = ksrc[j];
      const bf16* vsrc = Vh + (size_t)(kb * BC + r) * HD + c0;
#pragma unroll
      for (int j = 0; j < 64; ++j) Vt[c0 + j][r] = vsrc[j];
    }
    __syncthreads();

    // -- S = Q K^T : 4 n-tiles x 4 k-chunks = 16 WMMAs ---------------------
    v8f s[4] = {};
#pragma unroll
    for (int kc = 0; kc < 4; ++kc)
#pragma unroll
      for (int n = 0; n < 4; ++n) {
        v16bf b = load_b_frag_bf16(&Kt[n * 16 + lrow][kc * 32 + half * 16]);
        s[n] = wmma_bf16(aq[kc], b, s[n]);
      }

    // -- scale, causal mask, online softmax --------------------------------
#pragma unroll
    for (int r = 0; r < 8; ++r) {
      const int qg = qb * BR + wv * 16 + half * 8 + r;
      float mx = -INFINITY;
#pragma unroll
      for (int n = 0; n < 4; ++n) {
        const int kg = kb * BC + n * 16 + lrow;
        float v = s[n][r] * scale;
        v = (kg <= qg) ? v : -INFINITY;
        s[n][r] = v;
        mx = fmaxf(mx, v);
      }
#pragma unroll
      for (int msk = 1; msk < 16; msk <<= 1) mx = fmaxf(mx, __shfl_xor(mx, msk, 32));
      const float mnew = fmaxf(m[r], mx);
      const float alpha = __expf(m[r] - mnew);  // 0 on first block
      m[r] = mnew;
      l[r] *= alpha;
#pragma unroll
      for (int n2 = 0; n2 < 8; ++n2) o[n2][r] *= alpha;
      float rs = 0.0f;
#pragma unroll
      for (int n = 0; n < 4; ++n) {
        const float p = __expf(s[n][r] - mnew);  // masked -> exp(-inf)=0
        s[n][r] = p;
        rs += p;
      }
#pragma unroll
      for (int msk = 1; msk < 16; msk <<= 1) rs += __shfl_xor(rs, msk, 32);
      l[r] += rs;
    }

    // -- restage P (C layout) into A-fragment layout via LDS ---------------
#pragma unroll
    for (int n = 0; n < 4; ++n)
#pragma unroll
      for (int r = 0; r < 8; ++r)
        Ps[wv][half * 8 + r][n * 16 + lrow] = (bf16)s[n][r];
    // wave-private LDS RAW: hardware keeps DS ops in order; the wait + memory
    // clobber pins the compiler's ordering too.
    asm volatile("s_wait_dscnt 0" ::: "memory");

    // -- O += P V : 8 d-tiles x 2 k-chunks = 16 WMMAs ----------------------
#pragma unroll
    for (int kc = 0; kc < 2; ++kc) {
      v16bf ap = load_a_frag(&Ps[wv][0][0], BC + 8, lrow, kc * 32, half);
#pragma unroll
      for (int n = 0; n < 8; ++n) {
        v16bf bv = load_b_frag_bf16(&Vt[n * 16 + lrow][kc * 32 + half * 16]);
        o[n] = wmma_bf16(ap, bv, o[n]);
      }
    }
    __syncthreads();  // protect Kt/Vt before next iteration's load
  }

  // -- epilogue: y[t, h*128 + d] = o / l -----------------------------------
#pragma unroll
  for (int r = 0; r < 8; ++r) {
    const float inv = 1.0f / l[r];
    const int t = qb * BR + wv * 16 + half * 8 + r;
#pragma unroll
    for (int n = 0; n < 8; ++n)
      Y[(size_t)t * HTOT + h * HD + n * 16 + lrow] = (bf16)(o[n][r] * inv);
  }
}

// ---------------------------------------------------------------------------
// launch
// ---------------------------------------------------------------------------
extern "C" void kernel_launch(void* const* d_in, const int* in_sizes, int n_in,
                              void* d_out, int out_size, void* d_ws, size_t ws_size,
                              hipStream_t stream) {
  const float* x        = (const float*)d_in[0];
  const float* ve       = (const float*)d_in[1];
  const float* qkv_w    = (const float*)d_in[2];
  const float* lambdas  = (const float*)d_in[3];
  const float* c_proj_w = (const float*)d_in[4];
  float* out = (float*)d_out;

  char* ws = (char*)d_ws;
  bf16* qkv = (bf16*)ws; ws += (size_t)SEQ * QKVN * sizeof(bf16);
  bf16* Qb  = (bf16*)ws; ws += (size_t)SEQ * HTOT * sizeof(bf16);
  bf16* Kb  = (bf16*)ws; ws += (size_t)SEQ * HTOT * sizeof(bf16);
  bf16* Vb  = (bf16*)ws; ws += (size_t)SEQ * HTOT * sizeof(bf16);
  bf16* Yb  = (bf16*)ws; ws += (size_t)SEQ * HTOT * sizeof(bf16);

  // 1) qkv = x @ qkv_w^T   (M=4096, N=6144, K=2048) -> bf16
  gemm_wmma_kernel<float, true>
      <<<(SEQ / 128) * (QKVN / 128), 256, 0, stream>>>(x, qkv_w, qkv, SEQ, QKVN, DIMX);
  // 2) RMS-norm + RoPE + v mix -> Q/K/V (H,T,D) bf16
  norm_rope_mix_kernel<<<SEQ * NHEADS, 128, 0, stream>>>(qkv, ve, lambdas, Qb, Kb, Vb);
  // 3) causal flash attention -> Y (T, HTOT) bf16
  attention_kernel<<<(SEQ / BR) * NHEADS, 128, 0, stream>>>(Qb, Kb, Vb, Yb);
  // 4) out = Y @ c_proj_w^T (M=4096, N=2048, K=2048) -> fp32
  gemm_wmma_kernel<bf16, false>
      <<<(SEQ / 128) * (DIMX / 128), 256, 0, stream>>>(Yb, c_proj_w, out, SEQ, DIMX, HTOT);
}
